// Attention_35682588295380
// MI455X (gfx1250) — matrix operand order repaired and checked by reading, not compiled, not run on hardware
//
#include <hip/hip_runtime.h>
#include <cstdint>
#include <cstddef>

// ---------------------------------------------------------------------------
// Problem constants (match reference)
// ---------------------------------------------------------------------------
#define B_   2
#define S_   2048
#define D_   2048
#define H_   16
#define KV_  4
#define HD_  128
#define REP_ (H_ / KV_)

typedef unsigned short u16;
typedef __bf16 bf16_t;
typedef bf16_t v16bf __attribute__((ext_vector_type(16)));
typedef float  v8f   __attribute__((ext_vector_type(8)));

union Frag {
  uint4 u[2];
  v16bf v;
  u16   s[16];
};

__device__ __forceinline__ u16 f2bf(float f) {
  uint32_t u = __float_as_uint(f);
  uint32_t r = u + 0x7FFFu + ((u >> 16) & 1u);   // round-to-nearest-even
  return (u16)(r >> 16);
}
__device__ __forceinline__ float bf2f(u16 h) {
  return __uint_as_float(((uint32_t)h) << 16);
}
__device__ __forceinline__ void stor(u16* p, float v)   { *p = f2bf(v); }
__device__ __forceinline__ void stor(float* p, float v) { *p = v; }

__device__ __forceinline__ v8f wmma_bf16(const Frag& a, const Frag& b, v8f c) {
  return __builtin_amdgcn_wmma_f32_16x16x32_bf16(
      /*neg_a=*/false, a.v, /*neg_b=*/false, b.v,
      /*c_mod=*/(short)0, c, /*reuse_a=*/false, /*reuse_b=*/false);
}

// Async global -> LDS copy, 16B per lane (tracked by ASYNCcnt).
__device__ __forceinline__ void async_load_lds_b128(uint32_t lds_off, uint64_t gaddr) {
  asm volatile("global_load_async_to_lds_b128 %0, %1, off"
               :: "v"(lds_off), "v"(gaddr) : "memory");
}
__device__ __forceinline__ void wait_asynccnt0() {
  asm volatile("s_wait_asynccnt 0" ::: "memory");
}

// ---------------------------------------------------------------------------
// f32 -> bf16 convert (grid stride)
// ---------------------------------------------------------------------------
__global__ void f32_to_bf16_kernel(const float* __restrict__ s, u16* __restrict__ d, int n) {
  for (int i = blockIdx.x * blockDim.x + threadIdx.x; i < n; i += gridDim.x * blockDim.x)
    d[i] = f2bf(s[i]);
}

// ---------------------------------------------------------------------------
// NT GEMM: C[m][n] = sum_k A[m][k] * Bw[n][k]
// A: M x K bf16 row-major (K contiguous), Bw: N x K bf16 row-major.
// Block = 128 threads (4 waves). Block tile 64(M) x 128(N); wave tile 16 x 128
// (8 WMMAs share one A-fragment per K-step => 2x arithmetic intensity vs 16x64).
// k order used here (contiguous, not ISA table order): A-frag lane<16 holds K 0..15; lane>=16 K 16..31.
//                 B-frag lane<16 holds col n=lane, K 0..15; lane>=16 K 16..31.
// ---------------------------------------------------------------------------
template <typename OutT>
__global__ void __launch_bounds__(128)
gemm_nt_wmma(const u16* __restrict__ A, const u16* __restrict__ Bw,
             OutT* __restrict__ C, int M, int N, int K) {
  const int  lane = threadIdx.x & 31;
  const int  wave = threadIdx.x >> 5;
  const bool hi   = lane >= 16;
  const int  lm   = lane & 15;
  const int  m0   = blockIdx.y * 64 + wave * 16;
  const int  n0   = blockIdx.x * 128;

  v8f acc[8] = {};

  const uint4* Arow = (const uint4*)(A + (size_t)(m0 + lm) * K);
  const uint4* Brow[8];
#pragma unroll
  for (int t = 0; t < 8; ++t)
    Brow[t] = (const uint4*)(Bw + (size_t)(n0 + 16 * t + lm) * K);

  const int K4 = K >> 3;                   // row length in uint4 (8 halfs each)
  for (int k4 = 0; k4 < K4; k4 += 4) {     // 32 halfs per step
    Frag a;
    a.u[0] = Arow[k4 + (hi ? 2 : 0)];
    a.u[1] = Arow[k4 + (hi ? 3 : 1)];
    const int o = k4 + (hi ? 2 : 0);
#pragma unroll
    for (int t = 0; t < 8; ++t) {
      Frag b;
      b.u[0] = Brow[t][o];
      b.u[1] = Brow[t][o + 1];
      acc[t] = wmma_bf16(a, b, acc[t]);
    }
  }

#pragma unroll
  for (int t = 0; t < 8; ++t)
#pragma unroll
    for (int r = 0; r < 8; ++r) {
      const int row = m0 + r + (hi ? 8 : 0);
      const int col = n0 + 16 * t + lm;
      stor(&C[(size_t)row * N + col], acc[t][r]);
    }
}

// ---------------------------------------------------------------------------
// RoPE (computed analytically; matches reference rope_cos_sin/apply_rope)
// inv_freq[i] = 10000^(-(2i)/HD), pair (2i, 2i+1) rotated by angle pos*inv_freq.
// ---------------------------------------------------------------------------
#define LOG2_THETA 13.287712379549449f   // log2(10000)

__global__ void rope_q_kernel(const u16* __restrict__ qf, u16* __restrict__ qb, int total) {
  int idx = blockIdx.x * blockDim.x + threadIdx.x;
  if (idx >= total) return;
  const int i = idx & 63;
  const int h = (idx >> 6) & (H_ - 1);
  const int s = (idx >> 10) & (S_ - 1);
  const int b = idx >> 21;
  const float inv = exp2f(-(2.0f * (float)i / (float)HD_) * LOG2_THETA);
  const float f = (float)s * inv;
  const float c = cosf(f), sn = sinf(f);
  const size_t src = ((size_t)(b * S_ + s) * H_ + h) * HD_ + 2 * i;   // [b,s,h,d]
  const float x1 = bf2f(qf[src]), x2 = bf2f(qf[src + 1]);
  const size_t dst = ((size_t)(b * H_ + h) * S_ + s) * HD_ + 2 * i;   // [b,h,s,d]
  qb[dst]     = f2bf(x1 * c - x2 * sn);
  qb[dst + 1] = f2bf(x1 * sn + x2 * c);
}

__global__ void rope_k_kernel(const u16* __restrict__ kf, u16* __restrict__ kb, int total) {
  int idx = blockIdx.x * blockDim.x + threadIdx.x;
  if (idx >= total) return;
  const int i  = idx & 63;
  const int kv = (idx >> 6) & (KV_ - 1);
  const int s  = (idx >> 8) & (S_ - 1);
  const int b  = idx >> 19;
  const float inv = exp2f(-(2.0f * (float)i / (float)HD_) * LOG2_THETA);
  const float f = (float)s * inv;
  const float c = cosf(f), sn = sinf(f);
  const size_t src = ((size_t)(b * S_ + s) * KV_ + kv) * HD_ + 2 * i; // [b,s,kv,d]
  const float x1 = bf2f(kf[src]), x2 = bf2f(kf[src + 1]);
  const size_t dst = ((size_t)(b * KV_ + kv) * S_ + s) * HD_ + 2 * i; // [b,kv,s,d]
  kb[dst]     = f2bf(x1 * c - x2 * sn);
  kb[dst + 1] = f2bf(x1 * sn + x2 * c);
}

__global__ void relayout_v_kernel(const u16* __restrict__ vf, u16* __restrict__ vb, int total) {
  int idx = blockIdx.x * blockDim.x + threadIdx.x;
  if (idx >= total) return;
  const int d  = idx & (HD_ - 1);
  const int kv = (idx >> 7) & (KV_ - 1);
  const int s  = (idx >> 9) & (S_ - 1);
  const int b  = idx >> 20;
  vb[((size_t)(b * KV_ + kv) * S_ + s) * HD_ + d] =
      vf[((size_t)(b * S_ + s) * KV_ + kv) * HD_ + d];
}

// ---------------------------------------------------------------------------
// Flash attention (causal, ALiBi computed analytically).
// Grid: (S/64, H, B). Block = 128 threads = 4 waves; wave w owns q rows
// [qb0+16w, qb0+16w+16). Key tiles of 32, shared by all 4 waves via LDS.
// K tile is staged with GLOBAL_LOAD_ASYNC_TO_LDS_B128 (ASYNCcnt path);
// V tile is transposed on the way into LDS (manual, needs the shuffle).
// ---------------------------------------------------------------------------
__global__ void __launch_bounds__(128)
flash_attn_kernel(const u16* __restrict__ Q,   // [B][H][S][HD]
                  const u16* __restrict__ Kt,  // [B][KV][S][HD]
                  const u16* __restrict__ V,   // [B][KV][S][HD]
                  u16* __restrict__ O) {       // [B*S][H*HD]
  __shared__ __align__(16) u16 lds_k[32][136];      // [key][dim], padded
  __shared__ __align__(16) u16 lds_vt[HD_][40];     // [dim][key], padded
  __shared__ __align__(16) u16 lds_p[4][16][40];    // per-wave P tile

  const int  tid  = threadIdx.x;
  const int  lane = tid & 31;
  const int  wave = tid >> 5;
  const bool hi   = lane >= 16;
  const int  lm   = lane & 15;

  const int qb0 = blockIdx.x * 64;
  const int h   = blockIdx.y;
  const int b   = blockIdx.z;
  const int g   = h / REP_;                 // kv head
  const int qw0 = qb0 + wave * 16;

  const float slope  = exp2f(-8.0f * (float)(h + 1) / (float)H_);
  const float sscale = 0.08838834764831845f;     // 1/sqrt(HD)

  // Q A-fragments for this wave's 16 rows (4 chunks of K=32 dims)
  Frag aq[4];
  {
    const uint4* qrow =
        (const uint4*)(Q + ((size_t)(b * H_ + h) * S_ + qw0 + lm) * HD_);
#pragma unroll
    for (int c = 0; c < 4; ++c) {
      aq[c].u[0] = qrow[4 * c + (hi ? 2 : 0)];
      aq[c].u[1] = qrow[4 * c + (hi ? 3 : 1)];
    }
  }

  v8f   acc[8] = {};
  float m_run[8], l_run[8];
#pragma unroll
  for (int r = 0; r < 8; ++r) { m_run[r] = -1e30f; l_run[r] = 0.0f; }

  const u16* Kbase = Kt + (size_t)(b * KV_ + g) * S_ * HD_;
  const u16* Vbase = V  + (size_t)(b * KV_ + g) * S_ * HD_;

  const int kend = qb0 + 64;                // causal: keys < block's max q + 1
  for (int kt = 0; kt < kend; kt += 32) {
    // ---- K tile: async global->LDS DMA, 512 x 16B ----
#pragma unroll
    for (int i = 0; i < 4; ++i) {
      const int e = tid + 128 * i;
      const int key = e >> 4, seg = e & 15;
      const uint64_t gaddr =
          (uint64_t)(uintptr_t)(Kbase + (size_t)(kt + key) * HD_ + seg * 8);
      const uint32_t laddr = (uint32_t)(uintptr_t)&lds_k[key][seg * 8];
      async_load_lds_b128(laddr, gaddr);
    }
    // ---- V tile transposed into LDS (manual: layout change) ----
#pragma unroll
    for (int i = 0; i < 16; ++i) {          // 2048 dwords total
      const int e = tid + 128 * i;
      const int key = e >> 6, dp = e & 63;
      uint32_t u = ((const uint32_t*)(Vbase + (size_t)(kt + key) * HD_))[dp];
      lds_vt[2 * dp][key]     = (u16)(u & 0xFFFFu);
      lds_vt[2 * dp + 1][key] = (u16)(u >> 16);
    }
    wait_asynccnt0();
    __syncthreads();

    // ---- scores: s0 = keys [0,16), s1 = keys [16,32) ----
    v8f s0 = {}, s1 = {};
    {
      const uint4* k0p = (const uint4*)&lds_k[lm][0];
      const uint4* k1p = (const uint4*)&lds_k[16 + lm][0];
      const int doff = hi ? 2 : 0;
#pragma unroll
      for (int c = 0; c < 4; ++c) {
        Frag b0, b1;
        b0.u[0] = k0p[4 * c + doff]; b0.u[1] = k0p[4 * c + doff + 1];
        b1.u[0] = k1p[4 * c + doff]; b1.u[1] = k1p[4 * c + doff + 1];
        s0 = wmma_bf16(aq[c], b0, s0);
        s1 = wmma_bf16(aq[c], b1, s1);
      }
    }

    // ---- online softmax (rows split across lane halves per C layout) ----
#pragma unroll
    for (int r = 0; r < 8; ++r) {
      const int qpos = qw0 + r + (hi ? 8 : 0);
      const int k0 = kt + lm, k1 = kt + 16 + lm;
      float f0 = s0[r] * sscale + slope * (float)(k0 - qpos) + (k0 > qpos ? -1e9f : 0.0f);
      float f1 = s1[r] * sscale + slope * (float)(k1 - qpos) + (k1 > qpos ? -1e9f : 0.0f);

      float tm = fmaxf(f0, f1);
      tm = fmaxf(tm, __shfl_xor(tm, 1, 16));
      tm = fmaxf(tm, __shfl_xor(tm, 2, 16));
      tm = fmaxf(tm, __shfl_xor(tm, 4, 16));
      tm = fmaxf(tm, __shfl_xor(tm, 8, 16));

      const float nm = fmaxf(m_run[r], tm);
      const float al = __expf(m_run[r] - nm);
      const float p0 = __expf(f0 - nm);
      const float p1 = __expf(f1 - nm);

      float rs = p0 + p1;
      rs += __shfl_xor(rs, 1, 16);
      rs += __shfl_xor(rs, 2, 16);
      rs += __shfl_xor(rs, 4, 16);
      rs += __shfl_xor(rs, 8, 16);

      l_run[r] = l_run[r] * al + rs;
      m_run[r] = nm;
#pragma unroll
      for (int t = 0; t < 8; ++t) acc[t][r] *= al;

      const int prow = r + (hi ? 8 : 0);
      lds_p[wave][prow][lm]      = f2bf(p0);
      lds_p[wave][prow][16 + lm] = f2bf(p1);
    }
    __syncthreads();

    // ---- P as A-fragment, then PV over 8 HD column tiles ----
    Frag ap;
    {
      const u16* pr = &lds_p[wave][lm][0];
      ap.u[0] = *(const uint4*)(pr + (hi ? 16 : 0));
      ap.u[1] = *(const uint4*)(pr + (hi ? 24 : 8));
    }
#pragma unroll
    for (int t = 0; t < 8; ++t) {
      Frag bv;
      const uint4* vr = (const uint4*)&lds_vt[16 * t + lm][hi ? 16 : 0];
      bv.u[0] = vr[0];
      bv.u[1] = vr[1];
      acc[t] = wmma_bf16(ap, bv, acc[t]);
    }
    __syncthreads();
  }

  // ---- normalize + write [b,s,h,d] bf16 for the Wo GEMM ----
#pragma unroll
  for (int t = 0; t < 8; ++t)
#pragma unroll
    for (int r = 0; r < 8; ++r) {
      const int qpos = qw0 + r + (hi ? 8 : 0);
      const float v = acc[t][r] / l_run[r];
      O[(size_t)(b * S_ + qpos) * (H_ * HD_) + h * HD_ + 16 * t + lm] = f2bf(v);
    }
}

// ---------------------------------------------------------------------------
// Host-side orchestration
// ---------------------------------------------------------------------------
extern "C" void kernel_launch(void* const* d_in, const int* in_sizes, int n_in,
                              void* d_out, int out_size, void* d_ws, size_t ws_size,
                              hipStream_t stream) {
  (void)in_sizes; (void)n_in; (void)out_size; (void)ws_size;
  const float* x  = (const float*)d_in[0];
  // d_in[1]=mask, d_in[2]=alibi_bias: values are closed-form; computed on-chip.
  const float* wq = (const float*)d_in[3];
  const float* wk = (const float*)d_in[4];
  const float* wv = (const float*)d_in[5];
  const float* wo = (const float*)d_in[6];
  float* out = (float*)d_out;

  char*  ws  = (char*)d_ws;
  size_t off = 0;
  auto bump = [&](size_t bytes) -> void* {
    void* p = ws + off;
    off += (bytes + 255) & ~(size_t)255;
    return p;
  };

  const size_t M   = (size_t)B_ * S_;        // 4096 token rows
  const int    Dkv = KV_ * HD_;              // 512

  u16* xb    = (u16*)bump(M * D_ * 2);
  u16* wqb   = (u16*)bump((size_t)D_ * D_ * 2);
  u16* wkb   = (u16*)bump((size_t)Dkv * D_ * 2);
  u16* wvb   = (u16*)bump((size_t)Dkv * D_ * 2);
  u16* wob   = (u16*)bump((size_t)D_ * D_ * 2);
  u16* qf    = (u16*)bump(M * D_ * 2);       // [b,s,h,d] pre-RoPE
  u16* kf    = (u16*)bump(M * Dkv * 2);      // [b,s,kv,d] pre-RoPE
  u16* vf    = (u16*)bump(M * Dkv * 2);
  u16* qb    = (u16*)bump(M * D_ * 2);       // [b,h,s,d]
  u16* kb    = (u16*)bump(M * Dkv * 2);      // [b,kv,s,d]
  u16* vb    = (u16*)bump(M * Dkv * 2);      // [b,kv,s,d]
  u16* attnb = (u16*)bump(M * D_ * 2);       // [b,s,h*d]

  const dim3 cvt_grid(4096), cvt_blk(256);
  f32_to_bf16_kernel<<<cvt_grid, cvt_blk, 0, stream>>>(x,  xb,  (int)(M * D_));
  f32_to_bf16_kernel<<<cvt_grid, cvt_blk, 0, stream>>>(wq, wqb, D_ * D_);
  f32_to_bf16_kernel<<<cvt_grid, cvt_blk, 0, stream>>>(wk, wkb, Dkv * D_);
  f32_to_bf16_kernel<<<cvt_grid, cvt_blk, 0, stream>>>(wv, wvb, Dkv * D_);
  f32_to_bf16_kernel<<<cvt_grid, cvt_blk, 0, stream>>>(wo, wob, D_ * D_);

  // Projections: x @ W^T  (NT GEMM, both K-contiguous). Block tile 64 x 128.
  gemm_nt_wmma<u16><<<dim3(D_ / 128,  (int)M / 64), 128, 0, stream>>>(xb, wqb, qf, (int)M, D_,  D_);
  gemm_nt_wmma<u16><<<dim3(Dkv / 128, (int)M / 64), 128, 0, stream>>>(xb, wkb, kf, (int)M, Dkv, D_);
  gemm_nt_wmma<u16><<<dim3(Dkv / 128, (int)M / 64), 128, 0, stream>>>(xb, wvb, vf, (int)M, Dkv, D_);

  // RoPE + relayout
  {
    const int nq = B_ * S_ * H_ * (HD_ / 2);
    rope_q_kernel<<<nq / 256, 256, 0, stream>>>(qf, qb, nq);
    const int nk = B_ * S_ * KV_ * (HD_ / 2);
    rope_k_kernel<<<nk / 256, 256, 0, stream>>>(kf, kb, nk);
    const int nv = B_ * S_ * KV_ * HD_;
    relayout_v_kernel<<<nv / 256, 256, 0, stream>>>(vf, vb, nv);
  }

  // Attention (causal + analytic ALiBi), flash-style
  flash_attn_kernel<<<dim3(S_ / 64, H_, B_), 128, 0, stream>>>(qb, kb, vb, attnb);

  // Output projection: attn @ wo^T -> f32 out
  gemm_nt_wmma<float><<<dim3(D_ / 128, (int)M / 64), 128, 0, stream>>>(attnb, wob, out, (int)M, D_, D_);
}
